// VGG_SNN_STDB_9620726743648
// MI455X (gfx1250) — matrix operand order repaired and checked by reading, not compiled, not run on hardware
//
#include <hip/hip_runtime.h>

typedef _Float16 half_t;
typedef __attribute__((ext_vector_type(16))) _Float16 v16h;
typedef __attribute__((ext_vector_type(8)))  float    v8f;

#define THR 1.0f
#define TSTEPS 20

// ---------------- elementwise / state kernels ----------------

__global__ void zero_f32_kernel(float* p, int n) {
  int i = blockIdx.x * blockDim.x + threadIdx.x;
  if (i < n) p[i] = 0.0f;
}

// v = v + x - s_prev ; s = (v > 1)
__global__ void encoder_kernel(const float* __restrict__ x, float* v, float* s, int n) {
  int i = blockIdx.x * blockDim.x + threadIdx.x;
  if (i >= n) return;
  float vv = v[i] + x[i] - s[i];
  s[i] = vv > 1.0f ? 1.0f : 0.0f;
  v[i] = vv;
}

// Direct 3x3 conv (3->64, 32x32, SAME) fused with LIF. K=27 -> not worth WMMA.
__global__ void conv1_lif_kernel(const float* __restrict__ s, const float* __restrict__ W1,
                                 float* m1, half_t* o1) {
  int i = blockIdx.x * blockDim.x + threadIdx.x;  // (b, oc, y, x) NCHW, 16*64*32*32
  int x  = i & 31;
  int y  = (i >> 5) & 31;
  int oc = (i >> 10) & 63;
  int b  = i >> 16;
  float acc = 0.0f;
#pragma unroll
  for (int c = 0; c < 3; ++c) {
    const float* wp = W1 + (oc * 3 + c) * 9;
    const float* sp = s + ((b * 3 + c) * 32) * 32;
#pragma unroll
    for (int ky = 0; ky < 3; ++ky) {
      int yy = y + ky - 1;
      if (yy < 0 || yy > 31) continue;
#pragma unroll
      for (int kx = 0; kx < 3; ++kx) {
        int xx = x + kx - 1;
        if (xx < 0 || xx > 31) continue;
        acc += sp[yy * 32 + xx] * wp[ky * 3 + kx];
      }
    }
  }
  float m = m1[i];
  float spk = m > THR ? 1.0f : 0.0f;   // spike from pre-update membrane
  m1[i] = m + acc - spk;
  o1[i] = (half_t)spk;
}

// 2x2 avg pool, NCHW f16 -> NCHW f16
__global__ void pool_nchw_kernel(const half_t* __restrict__ in, half_t* out, int H, int n) {
  int i = blockIdx.x * blockDim.x + threadIdx.x;
  if (i >= n) return;
  int Ho = H >> 1;
  int x = i % Ho;
  int y = (i / Ho) % Ho;
  int bc = i / (Ho * Ho);
  const half_t* p = in + (bc * H + 2 * y) * H + 2 * x;
  float a = (float)p[0] + (float)p[1] + (float)p[H] + (float)p[H + 1];
  out[i] = (half_t)(a * 0.25f);
}

// ---------------- fragment-layout packing ----------------
// Packed-A layout: dst[((mt*kTiles + kt)*32 + lane)*16 + ii] = A[mt*16 + (lane&15)][kt*32 + kOff]
// kOff per ISA 16-bit A layout: lanes 0-15 hold K {0..7,16..23}; lanes 16-31 hold {8..15,24..31}.
__device__ __forceinline__ int frag_koff(int lane, int ii) {
  return ((lane >> 4) << 3) + (ii & 7) + ((ii & 8) << 1);
}

// im2col for 3x3 SAME conv, writing the packed-A fragment layout directly.
// k = c*9 + ky*3 + kx (matches [oc][c][ky][kx] weight flattening).
__global__ void im2col_pack_kernel(const half_t* __restrict__ src, half_t* dst,
                                   int Cin, int HW, int Ktot, int nhwc, int n) {
  int i = blockIdx.x * blockDim.x + threadIdx.x;
  if (i >= n) return;
  int ii = i & 15;
  int lane = (i >> 4) & 31;
  int t = i >> 9;
  int kTiles = Ktot >> 5;
  int kt = t % kTiles;
  int mt = t / kTiles;
  int m = (mt << 4) + (lane & 15);
  int k = (kt << 5) + frag_koff(lane, ii);
  int c = k / 9;
  int r = k - c * 9;
  int ky = r / 3, kx = r - ky * 3;
  int x0 = m % HW;
  int y0 = (m / HW) % HW;
  int b = m / (HW * HW);
  int yy = y0 + ky - 1, xx = x0 + kx - 1;
  half_t val = (half_t)0.0f;
  if (yy >= 0 && yy < HW && xx >= 0 && xx < HW) {
    val = nhwc ? src[((b * HW + yy) * HW + xx) * Cin + c]
               : src[((b * Cin + c) * HW + yy) * HW + xx];
  }
  dst[i] = val;
}

// Weights (N x K f32, row-major) -> fp16 packed-B fragment layout:
// dst[((kt*nTiles + nt)*32 + lane)*16 + ii] = W[nt*16 + (lane&15)][kt*32 + frag_koff(lane, ii)]
__global__ void pack_w_kernel(const float* __restrict__ W, half_t* dst, int N, int K, int n) {
  int i = blockIdx.x * blockDim.x + threadIdx.x;
  if (i >= n) return;
  int ii = i & 15;
  int lane = (i >> 4) & 31;
  int t = i >> 9;
  int nTiles = N >> 4;
  int nt = t % nTiles;
  int kt = t / nTiles;
  int k = (kt << 5) + frag_koff(lane, ii);
  int nn = (nt << 4) + (lane & 15);
  dst[i] = (half_t)W[nn * K + k];
}

// o3 NHWC (16,16,16,128) f16 -> 2x2 pool -> flatten k=c*64+y*8+x -> packed-A (M=16, K=8192)
__global__ void pool_flatten_pack_kernel(const half_t* __restrict__ src, half_t* dst, int n) {
  int i = blockIdx.x * blockDim.x + threadIdx.x;
  if (i >= n) return;
  int ii = i & 15;
  int lane = (i >> 4) & 31;
  int kt = i >> 9;            // single m-tile (M=16)
  int b = lane & 15;
  int k = (kt << 5) + frag_koff(lane, ii);
  int c = k >> 6;
  int r = k & 63;
  int py = r >> 3, px = r & 7;
  const half_t* p = src + ((b * 16 + 2 * py) * 16 + 2 * px) * 128 + c;
  float a = (float)p[0] + (float)p[128] + (float)p[16 * 128] + (float)p[16 * 128 + 128];
  dst[i] = (half_t)(a * 0.25f);
}

// ---------------- WMMA GEMM + fused LIF ----------------
// Each wave computes NT 16x16 tiles along N, reusing its A fragment NT times.
// Block = 64 threads (2 waves); block covers 2*NT*16 columns. Grid (M/16, N/(32*NT)).
template <int NT>
__global__ void gemm_lif_wmma_kernel(const half_t* __restrict__ Ap, const half_t* __restrict__ Bp,
                                     float* mem, half_t* out, int N, int K) {
  int lane = threadIdx.x & 31;
  int w = threadIdx.x >> 5;
  int mt = blockIdx.x;
  int ntBase = (blockIdx.y * 2 + w) * NT;
  int kTiles = K >> 5;
  int nTiles16 = N >> 4;
  const v16h* a_ptr = (const v16h*)Ap + (size_t)mt * kTiles * 32 + lane;
  const v16h* b_ptr = (const v16h*)Bp + (size_t)ntBase * 32 + lane;
  v8f z = {};
  v8f acc[NT];
#pragma unroll
  for (int j = 0; j < NT; ++j) acc[j] = z;
  for (int kt = 0; kt < kTiles; ++kt) {
    v16h a = a_ptr[(size_t)kt * 32];
    const v16h* bk = b_ptr + (size_t)kt * nTiles16 * 32;
    __builtin_prefetch((const void*)(bk + (size_t)nTiles16 * 32), 0, 3);  // near-scope
    __builtin_prefetch((const void*)(a_ptr + (size_t)(kt + 1) * 32), 0, 3);
#pragma unroll
    for (int j = 0; j < NT; ++j) {
      v16h b = bk[(size_t)j * 32];
      acc[j] = __builtin_amdgcn_wmma_f32_16x16x32_f16(
          false, a, false, b, (short)0, acc[j], false, false);
    }
  }
  // C layout: VGPR r -> row m0 + r + 8*(lane>=16), col n0 + (lane&15)
  int rowAdd = (lane >> 4) << 3;
  int col = lane & 15;
  int m0 = mt << 4;
#pragma unroll
  for (int j = 0; j < NT; ++j) {
    int n0 = (ntBase + j) << 4;
#pragma unroll
    for (int r = 0; r < 8; ++r) {
      size_t idx = (size_t)(m0 + r + rowAdd) * N + n0 + col;
      float mv = mem[idx];
      float spk = mv > THR ? 1.0f : 0.0f;
      mem[idx] = mv + acc[j][r] - spk;
      out[idx] = (half_t)spk;
    }
  }
}

// Split-K WMMA partial GEMM (no LIF): blockIdx.z selects the K slice.
// Writes f32 partials: part[(slice*16 + row)*N + col].
template <int NT>
__global__ void gemm_partial_wmma_kernel(const half_t* __restrict__ Ap, const half_t* __restrict__ Bp,
                                         float* part, int N, int kTilesPerSlice) {
  int lane = threadIdx.x & 31;
  int w = threadIdx.x >> 5;
  int mt = blockIdx.x;
  int ntBase = (blockIdx.y * 2 + w) * NT;
  int slice = blockIdx.z;
  int kt0 = slice * kTilesPerSlice;
  int nTiles16 = N >> 4;
  const v16h* a_ptr = (const v16h*)Ap + ((size_t)mt * gridDim.z * kTilesPerSlice + kt0) * 32 + lane;
  const v16h* b_ptr = (const v16h*)Bp + ((size_t)kt0 * nTiles16 + ntBase) * 32 + lane;
  v8f z = {};
  v8f acc[NT];
#pragma unroll
  for (int j = 0; j < NT; ++j) acc[j] = z;
  for (int kt = 0; kt < kTilesPerSlice; ++kt) {
    v16h a = a_ptr[(size_t)kt * 32];
    const v16h* bk = b_ptr + (size_t)kt * nTiles16 * 32;
    __builtin_prefetch((const void*)(bk + (size_t)nTiles16 * 32), 0, 3);
#pragma unroll
    for (int j = 0; j < NT; ++j) {
      v16h b = bk[(size_t)j * 32];
      acc[j] = __builtin_amdgcn_wmma_f32_16x16x32_f16(
          false, a, false, b, (short)0, acc[j], false, false);
    }
  }
  int rowAdd = (lane >> 4) << 3;
  int col = lane & 15;
  int m0 = mt << 4;
#pragma unroll
  for (int j = 0; j < NT; ++j) {
    int n0 = (ntBase + j) << 4;
#pragma unroll
    for (int r = 0; r < 8; ++r) {
      size_t idx = (size_t)(slice * 16 + m0 + r + rowAdd) * N + n0 + col;
      part[idx] = acc[j][r];
    }
  }
}

// Sum 8 split-K partials (16x1024), apply LIF against m4, write spikes directly
// in packed-A fragment layout for the fc2 GEMM.
__global__ void fc1_reduce_lif_pack_kernel(const float* __restrict__ part, float* mem,
                                           half_t* dstPacked, int n) {
  int i = blockIdx.x * blockDim.x + threadIdx.x;   // n = 16384
  if (i >= n) return;
  int ii = i & 15;
  int lane = (i >> 4) & 31;
  int kt = i >> 9;                 // M=16 -> single m-tile
  int m = lane & 15;
  int col = (kt << 5) + frag_koff(lane, ii);
  float acc = 0.0f;
#pragma unroll
  for (int s = 0; s < 8; ++s) acc += part[(size_t)(s * 16 + m) * 1024 + col];
  int idx = m * 1024 + col;
  float mv = mem[idx];
  float spk = mv > THR ? 1.0f : 0.0f;
  mem[idx] = mv + acc - spk;
  dstPacked[i] = (half_t)spk;
}

// mo[b][j] += sum_k o5[b][k] * L3[j][k]  (M=16, N=10 -> scalar)
__global__ void fc3_kernel(const half_t* __restrict__ o5, const float* __restrict__ L3, float* mo) {
  int i = blockIdx.x * blockDim.x + threadIdx.x;
  if (i >= 160) return;
  int b = i / 10, j = i - (i / 10) * 10;
  const half_t* a = o5 + b * 1024;
  const float* wp = L3 + j * 1024;
  float acc = 0.0f;
  for (int k = 0; k < 1024; ++k) acc += (float)a[k] * wp[k];
  mo[i] += acc;
}

// ---------------- host launch ----------------

extern "C" void kernel_launch(void* const* d_in, const int* in_sizes, int n_in,
                              void* d_out, int out_size, void* d_ws, size_t ws_size,
                              hipStream_t stream) {
  (void)in_sizes; (void)n_in; (void)out_size; (void)ws_size;
  const float* x  = (const float*)d_in[0];
  const float* W1 = (const float*)d_in[1];
  const float* W2 = (const float*)d_in[2];
  const float* W3 = (const float*)d_in[3];
  const float* L1 = (const float*)d_in[4];
  const float* L2 = (const float*)d_in[5];
  const float* L3 = (const float*)d_in[6];
  float* mo = (float*)d_out;

  size_t off = 0;
  char* base = (char*)d_ws;
  auto carve = [&](size_t elems, size_t esz) -> char* {
    char* p = base + off;
    off += elems * esz;
    off = (off + 255) & ~(size_t)255;
    return p;
  };

  // f32 membranes/state, contiguous for one-shot zeroing (all sizes multiple of 64)
  const int N_STATE = 49152 * 2 + 1048576 + 524288 * 2 + 16384 * 2;  // 2,228,224
  float* stateF32 = (float*)carve(N_STATE, 4);
  float* v  = stateF32;
  float* s  = v + 49152;
  float* m1 = s + 49152;
  float* m2 = m1 + 1048576;
  float* m3 = m2 + 524288;
  float* m4 = m3 + 524288;
  float* m5 = m4 + 16384;

  float*  fc1part = (float*)carve(8 * 16 * 1024, 4);  // split-K partials
  half_t* o1   = (half_t*)carve(1048576, 2);   // conv1 spikes NCHW
  half_t* s1p  = (half_t*)carve(262144, 2);    // pooled spikes NCHW (16,64,16,16)
  half_t* col2 = (half_t*)carve(2359296, 2);   // packed-A, M=4096 K=576
  half_t* o2   = (half_t*)carve(524288, 2);    // conv2 spikes, row-major [m][c] == NHWC
  half_t* col3 = (half_t*)carve(4718592, 2);   // packed-A, M=4096 K=1152
  half_t* o3   = (half_t*)carve(524288, 2);    // conv3 spikes NHWC
  half_t* fp   = (half_t*)carve(131072, 2);    // packed-A, M=16 K=8192
  half_t* o4p  = (half_t*)carve(16384, 2);     // fc1 spikes, packed-A M=16 K=1024
  half_t* o5   = (half_t*)carve(16384, 2);     // fc2 spikes row-major
  half_t* W2p  = (half_t*)carve(73728, 2);     // packed-B 576x128
  half_t* W3p  = (half_t*)carve(147456, 2);    // packed-B 1152x128
  half_t* L1p  = (half_t*)carve(8388608, 2);   // packed-B 8192x1024
  half_t* L2p  = (half_t*)carve(1048576, 2);   // packed-B 1024x1024

  auto g = [](int n) { return dim3((unsigned)((n + 255) / 256)); };
  dim3 b256(256), b64(64);

  zero_f32_kernel<<<g(N_STATE), b256, 0, stream>>>(stateF32, N_STATE);
  zero_f32_kernel<<<g(160), b256, 0, stream>>>(mo, 160);

  // One-time weight conversion+swizzle (weights are timestep-invariant, live in L2)
  pack_w_kernel<<<g(73728),   b256, 0, stream>>>(W2, W2p, 128, 576, 73728);
  pack_w_kernel<<<g(147456),  b256, 0, stream>>>(W3, W3p, 128, 1152, 147456);
  pack_w_kernel<<<g(8388608), b256, 0, stream>>>(L1, L1p, 1024, 8192, 8388608);
  pack_w_kernel<<<g(1048576), b256, 0, stream>>>(L2, L2p, 1024, 1024, 1048576);

  for (int t = 0; t < TSTEPS; ++t) {
    encoder_kernel<<<g(49152), b256, 0, stream>>>(x, v, s, 49152);
    conv1_lif_kernel<<<g(1048576), b256, 0, stream>>>(s, W1, m1, o1);
    pool_nchw_kernel<<<g(262144), b256, 0, stream>>>(o1, s1p, 32, 262144);
    im2col_pack_kernel<<<g(2359296), b256, 0, stream>>>(s1p, col2, 64, 16, 576, 0, 2359296);
    // conv2: M=4096 N=128 K=576; block covers 128 cols
    gemm_lif_wmma_kernel<4><<<dim3(256, 1), b64, 0, stream>>>(col2, W2p, m2, o2, 128, 576);
    im2col_pack_kernel<<<g(4718592), b256, 0, stream>>>(o2, col3, 128, 16, 1152, 1, 4718592);
    // conv3: M=4096 N=128 K=1152
    gemm_lif_wmma_kernel<4><<<dim3(256, 1), b64, 0, stream>>>(col3, W3p, m3, o3, 128, 1152);
    pool_flatten_pack_kernel<<<g(131072), b256, 0, stream>>>(o3, fp, 131072);
    // fc1: M=16 N=1024 K=8192, split-K into 8 slices of 32 k-tiles
    gemm_partial_wmma_kernel<4><<<dim3(1, 8, 8), b64, 0, stream>>>(fp, L1p, fc1part, 1024, 32);
    fc1_reduce_lif_pack_kernel<<<g(16384), b256, 0, stream>>>(fc1part, m4, o4p, 16384);
    // fc2: M=16 N=1024 K=1024
    gemm_lif_wmma_kernel<4><<<dim3(1, 8), b64, 0, stream>>>(o4p, L2p, m5, o5, 1024, 1024);
    fc3_kernel<<<g(160), b256, 0, stream>>>(o5, L3, mo);
  }
}
